// TurboQuantizer_84533546320350
// MI455X (gfx1250) — matrix operand order repaired and checked by reading, not compiled, not run on hardware
//
#include <hip/hip_runtime.h>
#include <stdint.h>

typedef __attribute__((ext_vector_type(16))) _Float16 v16h;
typedef __attribute__((ext_vector_type(8)))  float    v8f;
typedef __attribute__((ext_vector_type(4)))  int      v4i;

#define HEAD_DIM 128
#define PSTR 130   // LDS row stride (halves) for Pi: 260B = 65 banks -> conflict-free
#define XSTR 130   // LDS row stride (halves) for 16-row tiles
#define EPS  1e-8f

#if defined(__has_builtin)
# if __has_builtin(__builtin_amdgcn_global_load_async_to_lds_b128)
#  define HAS_ASYNC_LDS 1
# endif
#endif
#ifndef HAS_ASYNC_LDS
# define HAS_ASYNC_LDS 0
#endif

#if HAS_ASYNC_LDS
// Builtin signature (recovered from the round-2/3 diagnostics):
//   (v4i addrspace(1)* src, v4i addrspace(3)* dst, imm int offset, imm int cpol)
typedef __attribute__((address_space(1))) v4i as1_v4i;
typedef __attribute__((address_space(3))) v4i as3_v4i;
#endif

__launch_bounds__(256)
__global__ void turboquant_kernel(const float* __restrict__ x,
                                  const float* __restrict__ Pi,
                                  const float* __restrict__ cent,
                                  uint8_t*  __restrict__ idx_out,
                                  _Float16* __restrict__ vn_out,
                                  _Float16* __restrict__ rn_out,
                                  int n_tiles)
{
    __shared__ _Float16 sh_pi[HEAD_DIM * PSTR];   // 33280 B, Pi row-major f16
    __shared__ _Float16 sh_xhat[16 * XSTR];       //  4160 B, normalized tile
    __shared__ _Float16 sh_a2[16 * XSTR];         //  4160 B, reconstructed tile
    __shared__ uint8_t  sh_idx[16 * HEAD_DIM];    //  2048 B
    __shared__ float    sh_res[16];               //    64 B
#if HAS_ASYNC_LDS
    __shared__ float    sh_x[2][16 * HEAD_DIM];   // 16384 B double-buffered x stage
#endif

    const int t       = threadIdx.x;       // 0..255, 8 waves of 32
    const int lane    = t & 31;
    const int wave    = t >> 5;            // 0..7 -> one 16-col tile per wave
    const int half_id = (lane >> 4) & 1;
    const int l16     = lane & 15;

    // Uniform-grid centroids: nearest = clamp(round((v - c0)/step), 0, 7)
    const float c0       = cent[0];
    const float step     = cent[1] - cent[0];
    const float inv_step = 1.0f / step;

    // Stage Pi (f32 global -> f16 LDS, row-major, padded stride), once per block.
    for (int i = t; i < HEAD_DIM * HEAD_DIM; i += 256) {
        int j = i >> 7, k = i & 127;
        sh_pi[j * PSTR + k] = (_Float16)Pi[i];
    }

#if HAS_ASYNC_LDS
    // Prime the async pipeline: stream the first tile's 8KB x-slab into LDS.
    // Thread t owns bytes [32t, 32t+32): two async b128 transfers.
    int buf = 0;
    {
        const float* s = x + (size_t)blockIdx.x * 16 * HEAD_DIM + t * 8;
        as3_v4i* d = (as3_v4i*)&sh_x[0][t * 8];
        __builtin_amdgcn_global_load_async_to_lds_b128((as1_v4i*)s, d, 0, 0);
        __builtin_amdgcn_global_load_async_to_lds_b128((as1_v4i*)s, d, 16, 0);
    }
#endif
    __syncthreads();

    for (int tile = blockIdx.x; tile < n_tiles; tile += gridDim.x) {
        const int R = tile * 16;

#if HAS_ASYNC_LDS
        // Issue next tile's async copy into the other buffer, then retire the
        // current tile's two transfers (async loads complete in order, so
        // waiting asynccnt<=2 drains the older pair). Thread t consumes only
        // the bytes it issued -> wave-local dependency, no barrier required.
        {
            const int nxt = tile + (int)gridDim.x;
            if (nxt < n_tiles) {
                const float* s = x + (size_t)nxt * 16 * HEAD_DIM + t * 8;
                as3_v4i* d = (as3_v4i*)&sh_x[buf ^ 1][t * 8];
                __builtin_amdgcn_global_load_async_to_lds_b128((as1_v4i*)s, d, 0, 0);
                __builtin_amdgcn_global_load_async_to_lds_b128((as1_v4i*)s, d, 16, 0);
                asm volatile("s_wait_asynccnt 0x2" ::: "memory");
            } else {
                asm volatile("s_wait_asynccnt 0x0" ::: "memory");
            }
        }
#endif

        // ---- Phase A: row norms + normalized f16 tile ----
        {
            const int row = t >> 4;           // 0..15
            const int cg  = (t & 15) * 8;     // 8 floats per thread per row
#if HAS_ASYNC_LDS
            const float* xr = &sh_x[buf][t * 8];   // == [row][cg], staged async
#else
            const float* xr = x + (size_t)(R + row) * HEAD_DIM + cg;
#endif
            float v[8];
            #pragma unroll
            for (int i = 0; i < 8; ++i) v[i] = xr[i];
            float s = 0.f;
            #pragma unroll
            for (int i = 0; i < 8; ++i) s += v[i] * v[i];
            // reduce across the 16 lanes sharing this row (wave32 shuffles)
            s += __shfl_xor(s, 1, 32);
            s += __shfl_xor(s, 2, 32);
            s += __shfl_xor(s, 4, 32);
            s += __shfl_xor(s, 8, 32);
            const float nrm = sqrtf(s);
            const float inv = 1.0f / (nrm + EPS);
            if ((t & 15) == 0) vn_out[R + row] = (_Float16)nrm;
            #pragma unroll
            for (int i = 0; i < 8; ++i)
                sh_xhat[row * XSTR + cg + i] = (_Float16)(v[i] * inv);
            if (t < 16) sh_res[t] = 0.f;
#if !HAS_ASYNC_LDS
            const int nxt = tile + (int)gridDim.x;
            if (nxt < n_tiles)
                __builtin_prefetch(x + (size_t)nxt * 16 * HEAD_DIM + t * 8, 0, 1);
#endif
        }
        __syncthreads();

        // ---- Phase B: rotated = x_hat @ Pi^T via WMMA f16; scalar quantize ----
        {
            const int n0 = wave * 16;   // this wave's 16 output columns
            v8f c = {};
            #pragma unroll
            for (int kk = 0; kk < 4; ++kk) {   // K = 128 in 4 steps of 32
                const int k0 = kk * 32;
                v16h a, b;
                #pragma unroll
                for (int v = 0; v < 8; ++v) {
                    // A 16x32 f16 layout: v<4 -> K=2v, v>=4 -> 16+2(v-4); +8*laneHalf
                    const int ka = k0 + ((v < 4) ? 2 * v : 16 + 2 * (v - 4)) + half_id * 8;
                    a[2 * v]     = sh_xhat[l16 * XSTR + ka];
                    a[2 * v + 1] = sh_xhat[l16 * XSTR + ka + 1];
                    // B 32x16 f16 layout: same K order as A
                    const int kb = k0 + ((v < 4) ? 2 * v : 16 + 2 * (v - 4)) + half_id * 8;
                    // B1[k][n] = Pi[n][k] -> contiguous along k in sh_pi row n
                    b[2 * v]     = sh_pi[(n0 + l16) * PSTR + kb];
                    b[2 * v + 1] = sh_pi[(n0 + l16) * PSTR + kb + 1];
                }
                c = __builtin_amdgcn_wmma_f32_16x16x32_f16(false, a, false, b,
                                                           (short)0, c, false, false);
            }
            // D layout: vgpr r, lane l -> (M = r + 8*laneHalf, N = n0 + (l&15))
            #pragma unroll
            for (int r = 0; r < 8; ++r) {
                const int m = r + half_id * 8;
                const int n = n0 + l16;
                float f = rintf((c[r] - c0) * inv_step);
                f = fminf(fmaxf(f, 0.0f), 7.0f);
                sh_idx[m * HEAD_DIM + n] = (uint8_t)(int)f;
                sh_a2[m * XSTR + n]      = (_Float16)(c0 + f * step);
            }
        }
        __syncthreads();

        // ---- Phase C: x_mse = recon @ Pi via WMMA; residual norm ----
        {
            const int k0o = wave * 16;  // this wave's 16 output columns (over k)
            v8f c = {};
            #pragma unroll
            for (int jj = 0; jj < 4; ++jj) {   // contraction over j = 128
                const int j0 = jj * 32;
                v16h a, b;
                #pragma unroll
                for (int v = 0; v < 8; ++v) {
                    const int ja = j0 + ((v < 4) ? 2 * v : 16 + 2 * (v - 4)) + half_id * 8;
                    a[2 * v]     = sh_a2[l16 * XSTR + ja];
                    a[2 * v + 1] = sh_a2[l16 * XSTR + ja + 1];
                    const int jb = j0 + ((v < 4) ? 2 * v : 16 + 2 * (v - 4)) + half_id * 8;
                    // B2[j][k] = Pi[j][k] -> strided along j in sh_pi
                    b[2 * v]     = sh_pi[jb * PSTR + (k0o + l16)];
                    b[2 * v + 1] = sh_pi[(jb + 1) * PSTR + (k0o + l16)];
                }
                c = __builtin_amdgcn_wmma_f32_16x16x32_f16(false, a, false, b,
                                                           (short)0, c, false, false);
            }
            #pragma unroll
            for (int r = 0; r < 8; ++r) {
                const int m  = r + half_id * 8;
                const int k  = k0o + l16;
                const float xh = (float)sh_xhat[m * XSTR + k];
                const float dv = xh - c[r];
                atomicAdd(&sh_res[m], dv * dv);    // ds_add_f32
            }
        }
        // ---- flush indices (sh_idx finalized at the phase-B barrier) ----
        {
            const int row = t >> 4;
            const int cg  = (t & 15) * 8;
            const uint64_t v = *(const uint64_t*)&sh_idx[row * HEAD_DIM + cg];
            *(uint64_t*)(idx_out + (size_t)(R + row) * HEAD_DIM + cg) = v;
        }
        __syncthreads();
        if (t < 16) rn_out[R + t] = (_Float16)sqrtf(sh_res[t]);
        // sh_res re-zero happens in next iter's Phase A by the same t<16 threads,
        // and all other touches are fenced by the Phase-A barrier.
#if HAS_ASYNC_LDS
        buf ^= 1;
#endif
    }
}

extern "C" void kernel_launch(void* const* d_in, const int* in_sizes, int n_in,
                              void* d_out, int out_size, void* d_ws, size_t ws_size,
                              hipStream_t stream) {
    const float* x    = (const float*)d_in[0];   // [1,32,4096,128] f32
    const float* Pi   = (const float*)d_in[1];   // [128,128] f32
    const float* cent = (const float*)d_in[2];   // [8] f32

    const int N       = in_sizes[0] / HEAD_DIM;  // 131072 vectors
    const int n_tiles = N / 16;                  // 8192 tiles of 16 rows

    // Output packing (tuple return, flat in return order):
    //   [N*128] uint8 mse_indices, then [N] f16 vec_norm, then [N] f16 res_norm
    uint8_t*  idx_out = (uint8_t*)d_out;
    _Float16* vn_out  = (_Float16*)(idx_out + (size_t)N * HEAD_DIM);
    _Float16* rn_out  = vn_out + N;

    int blocks = n_tiles < 2048 ? n_tiles : 2048; // persistent blocks amortize Pi staging
    turboquant_kernel<<<blocks, 256, 0, stream>>>(x, Pi, cent, idx_out, vn_out, rn_out,
                                                  n_tiles);
}
